// MultiheadAttention_22763326668803
// MI455X (gfx1250) — matrix operand order repaired and checked by reading, not compiled, not run on hardware
//
#include <hip/hip_runtime.h>
#include <hip/hip_bf16.h>

// Problem constants (from reference): E=1024, H=16, HD=64, T=1024, B=4, S=1024
// Rows of flattened (T,B,E)/(S,B,E) tensors: R = 4096, row index = t*B + b.

typedef _Float16 h8    __attribute__((ext_vector_type(8)));
typedef _Float16 v16h  __attribute__((ext_vector_type(16)));
typedef float    v8f   __attribute__((ext_vector_type(8)));

__device__ __forceinline__ v8f wmma_f16(v16h a, v16h b, v8f c) {
  // v_wmma_f32_16x16x32_f16  (D = A x B + C), f32 accumulate
  return __builtin_amdgcn_wmma_f32_16x16x32_f16(false, a, false, b, (short)0, c,
                                                false, false);
}

// ---- CDNA5 async global->LDS copy (ASYNCcnt path, CDNA5 ISA)
// GVS addressing: mem = SADDR(64b SGPR) + VADDR(32b VGPR) + IOFFSET.
// VDST VGPR carries the LDS byte offset (addrspace(3) offset == low 32 bits of
// the generic pointer).
__device__ __forceinline__ void async_ld_b128(unsigned lds_off, unsigned voff,
                                              unsigned long long base) {
  asm volatile("global_load_async_to_lds_b128 %0, %1, %2"
               :: "v"(lds_off), "v"(voff), "s"(base) : "memory");
}
__device__ __forceinline__ void wait_async0() {
  asm volatile("s_wait_asynccnt 0" ::: "memory");
}

// A fragment: 16x32 f16, LDS tile row-major [m][k], strideH multiple of 8 halves.
// Lane L: m = L&15, half = L>>4. Per ISA table the 16 halves are
//   K = 8*half .. 8*half+7   and   K = 16+8*half .. 23+8*half  -> two b128 loads.
__device__ __forceinline__ v16h load_a_frag(const _Float16* base, int strideH, int lane) {
  const int m = lane & 15, hf = (lane >> 4) & 1;
  h8 c0 = *(const h8*)(base + m * strideH + 8 * hf);
  h8 c1 = *(const h8*)(base + m * strideH + 16 + 8 * hf);
  v16h f;
#pragma unroll
  for (int i = 0; i < 8; ++i) { f[i] = c0[i]; f[8 + i] = c1[i]; }
  return f;
}

// B fragment: 32x16 f16 (KxN), LDS tile stored [n][k] (k contiguous),
// strideH multiple of 8 halves. Lane L: n = L&15, half = L>>4,
// K = 8*half .. 8*half+7 and K = 16+8*half .. 23+8*half (same as A).
__device__ __forceinline__ v16h load_b_frag(const _Float16* base, int strideH, int lane) {
  const int n = lane & 15, hf = (lane >> 4) & 1;
  h8 c0 = *(const h8*)(base + n * strideH + 8 * hf);
  h8 c1 = *(const h8*)(base + n * strideH + 16 + 8 * hf);
  v16h f;
#pragma unroll
  for (int i = 0; i < 8; ++i) { f[i] = c0[i]; f[8 + i] = c1[i]; }
  return f;
}

// ---------------------------------------------------------------------------
// Kernel 1: projection GEMM.  OUT(f16)[4096,1024] = ((X f32 @ Wsec f32) + bias)*scale
// Wsec used untransposed: B[k,n] = W[k,n]  (reference does X @ W[:E]).
// Both tiles need f32->f16 conversion, so staging stays on the VALU path.
__global__ __launch_bounds__(128) void proj_gemm_kernel(
    const float* __restrict__ X, const float* __restrict__ W,
    const float* __restrict__ bias, _Float16* __restrict__ OUT, float scale) {
  __shared__ __align__(16) _Float16 As[64 * 72];  // [m][k], pad stride 72
  __shared__ __align__(16) _Float16 Bs[64 * 40];  // [n][k], stride 40
  const int tid = threadIdx.x, lane = tid & 31, wv = tid >> 5;
  const int row0 = blockIdx.x * 64, col0 = blockIdx.y * 64;
  v8f acc[4] = {};
  for (int k0 = 0; k0 < 1024; k0 += 32) {
    if (k0 + 32 < 1024)  // hint next tiles into cache (global_prefetch)
      __builtin_prefetch(X + (size_t)(row0 + (tid >> 1)) * 1024 + k0 + 32 + (tid & 1) * 16, 0, 1);
#pragma unroll 1
    for (int idx = tid; idx < 64 * 8; idx += 128) {  // A: f32 -> f16
      int r = idx >> 3, c = (idx & 7) * 4;
      const float4 v = *(const float4*)(X + (size_t)(row0 + r) * 1024 + k0 + c);
      As[r * 72 + c + 0] = (_Float16)v.x;
      As[r * 72 + c + 1] = (_Float16)v.y;
      As[r * 72 + c + 2] = (_Float16)v.z;
      As[r * 72 + c + 3] = (_Float16)v.w;
    }
#pragma unroll 1
    for (int idx = tid; idx < 64 * 32; idx += 128) {  // B transposed stage
      int n = idx & 63, k = idx >> 6;                 // n fastest -> coalesced
      Bs[n * 40 + k] = (_Float16)W[(size_t)(k0 + k) * 1024 + col0 + n];
    }
    __syncthreads();
    v16h af = load_a_frag(As + wv * 16 * 72, 72, lane);
#pragma unroll
    for (int nt = 0; nt < 4; ++nt)
      acc[nt] = wmma_f16(af, load_b_frag(Bs + nt * 16 * 40, 40, lane), acc[nt]);
    __syncthreads();
  }
  const int n_lo = lane & 15, hf = lane >> 4;
#pragma unroll
  for (int nt = 0; nt < 4; ++nt) {
    const int col = col0 + nt * 16 + n_lo;
    const float bv = bias[col];
#pragma unroll
    for (int r = 0; r < 8; ++r) {
      const int row = row0 + wv * 16 + r + 8 * hf;
      OUT[(size_t)row * 1024 + col] = (_Float16)((acc[nt][r] + bv) * scale);
    }
  }
}

// ---------------------------------------------------------------------------
// Kernel 2: avg_w[b,t,s] = (q_scaled[t,b,:] . k[s,b,:]) / H   (f32 out)
// Pure-f16 operands: tiles staged with async global->LDS copies, K-tile 64.
__global__ __launch_bounds__(128) void avgw_gemm_kernel(
    const _Float16* __restrict__ Q, const _Float16* __restrict__ K,
    float* __restrict__ AVG) {
  __shared__ __align__(16) _Float16 As[64 * 72];
  __shared__ __align__(16) _Float16 Bs[64 * 72];
  const int tid = threadIdx.x, lane = tid & 31, wv = tid >> 5;
  const int row0 = blockIdx.x * 64, col0 = blockIdx.y * 64, b = blockIdx.z;
  const unsigned long long qb = (unsigned long long)(uintptr_t)Q;
  const unsigned long long kb = (unsigned long long)(uintptr_t)K;
  const unsigned as_base = (unsigned)(uintptr_t)As;
  const unsigned bs_base = (unsigned)(uintptr_t)Bs;
  v8f acc[4] = {};
  for (int k0 = 0; k0 < 1024; k0 += 64) {
#pragma unroll 1
    for (int idx = tid; idx < 64 * 8; idx += 128) {   // 16B chunks, async path
      int r = idx >> 3, c = (idx & 7) * 8;            // c in halves
      unsigned l = (unsigned)((r * 72 + c) * 2);
      unsigned vq = (unsigned)(((((row0 + r) * 4 + b) * 1024) + k0 + c) * 2);
      unsigned vk = (unsigned)(((((col0 + r) * 4 + b) * 1024) + k0 + c) * 2);
      async_ld_b128(as_base + l, vq, qb);
      async_ld_b128(bs_base + l, vk, kb);
    }
    wait_async0();
    __syncthreads();
#pragma unroll
    for (int d = 0; d < 64; d += 32) {
      v16h af = load_a_frag(As + wv * 16 * 72 + d, 72, lane);
#pragma unroll
      for (int nt = 0; nt < 4; ++nt)
        acc[nt] = wmma_f16(af, load_b_frag(Bs + nt * 16 * 72 + d, 72, lane), acc[nt]);
    }
    __syncthreads();
  }
  const int n_lo = lane & 15, hf = lane >> 4;
#pragma unroll
  for (int nt = 0; nt < 4; ++nt) {
    const int col = col0 + nt * 16 + n_lo;
#pragma unroll
    for (int r = 0; r < 8; ++r) {
      const int row = row0 + wv * 16 + r + 8 * hf;
      AVG[(size_t)b * 1024 * 1024 + (size_t)row * 1024 + col] = acc[nt][r] * (1.0f / 16.0f);
    }
  }
}

// ---------------------------------------------------------------------------
// Kernel 3: flash attention with dual (pos/neg) online softmax.
// Block = (b,h,qtile): 4 waves, each wave owns 16 query rows (64/WG).
// Q/K tiles staged asynchronously; V tiles need a transpose so stay on VALU/DS.
__global__ __launch_bounds__(128) void attn_kernel(
    const _Float16* __restrict__ Q, const _Float16* __restrict__ K,
    const _Float16* __restrict__ V, _Float16* __restrict__ OP) {
  __shared__ __align__(16) _Float16 Qs[64 * 72];       // [row][d], 64x64
  __shared__ __align__(16) _Float16 Ks[32 * 72];       // [s][d] == B-tile [n][k]
  __shared__ __align__(16) _Float16 VcT[32 * 40];      // [j][s]  B-tile for P@Vc
  __shared__ __align__(16) _Float16 VdT[32 * 40];      // [j][s]  B-tile for P@Vd
  __shared__ __align__(16) _Float16 Pp[4][16 * 32];    // per-wave P scratch (A layout)
  __shared__ __align__(16) _Float16 Pn[4][16 * 32];
  const int tid = threadIdx.x, lane = tid & 31, wv = tid >> 5;
  const int qt = blockIdx.x;
  const int b = blockIdx.y >> 4, h = blockIdx.y & 15;
  const unsigned long long qbase = (unsigned long long)(uintptr_t)Q;
  const unsigned long long kbase = (unsigned long long)(uintptr_t)K;
  const unsigned qs_base = (unsigned)(uintptr_t)Qs;
  const unsigned ks_base = (unsigned)(uintptr_t)Ks;
#pragma unroll 1
  for (int idx = tid; idx < 64 * 8; idx += 128) {      // stage Q tile once (async)
    int r = idx >> 3, c = (idx & 7) * 8;
    unsigned vq = (unsigned)(((((qt * 64 + r) * 4 + b) * 1024) + h * 64 + c) * 2);
    async_ld_b128(qs_base + (unsigned)((r * 72 + c) * 2), vq, qbase);
  }
  v8f aP0 = {}, aP1 = {}, aN0 = {}, aN1 = {};
  float mp[8], lp[8], mn[8], ln[8];
#pragma unroll
  for (int r = 0; r < 8; ++r) { mp[r] = -1e30f; mn[r] = -1e30f; lp[r] = 0.f; ln[r] = 0.f; }
  const int n_lo = lane & 15, hf = lane >> 4;
  for (int s0 = 0; s0 < 1024; s0 += 32) {
    __syncthreads();  // previous chunk's LDS reads done before restage
#pragma unroll 1
    for (int idx = tid; idx < 32 * 8; idx += 128) {    // K tile: async
      int r = idx >> 3, c = (idx & 7) * 8;
      unsigned vk = (unsigned)(((((s0 + r) * 4 + b) * 1024) + h * 64 + c) * 2);
      async_ld_b128(ks_base + (unsigned)((r * 72 + c) * 2), vk, kbase);
    }
#pragma unroll 1
    for (int idx = tid; idx < 1024; idx += 128) {      // V staged transposed
      int j = idx & 31, sc = idx >> 5;
      const _Float16* vrow = V + ((size_t)(s0 + sc) * 4 + b) * 1024;
      VcT[j * 40 + sc] = vrow[h * 32 + j];
      VdT[j * 40 + sc] = vrow[512 + h * 32 + j];
    }
    wait_async0();
    __syncthreads();
    v8f s0a = {}, s1a = {};                            // scores 16x32 (2 n-tiles)
#pragma unroll
    for (int d = 0; d < 64; d += 32) {
      v16h af = load_a_frag(Qs + wv * 16 * 72 + d, 72, lane);
      s0a = wmma_f16(af, load_b_frag(Ks + d, 72, lane), s0a);
      s1a = wmma_f16(af, load_b_frag(Ks + 16 * 72 + d, 72, lane), s1a);
    }
    _Float16* pp = Pp[wv];
    _Float16* pn = Pn[wv];
#pragma unroll
    for (int r = 0; r < 8; ++r) {  // dual online softmax, rows r+8*hf
      float x0 = s0a[r], x1 = s1a[r];
      float mx = fmaxf(x0, x1), mi = fminf(x0, x1);
#pragma unroll
      for (int o = 1; o < 16; o <<= 1) {               // stays in 16-lane group
        mx = fmaxf(mx, __shfl_xor(mx, o));
        mi = fminf(mi, __shfl_xor(mi, o));
      }
      float mpn = fmaxf(mp[r], mx);
      float al = __expf(mp[r] - mpn);
      float p0 = __expf(x0 - mpn), p1 = __expf(x1 - mpn);
      float rs = p0 + p1;
      float mnn = fmaxf(mn[r], -mi);                   // max of -scores
      float be = __expf(mn[r] - mnn);
      float q0 = __expf(-x0 - mnn), q1 = __expf(-x1 - mnn);
      float rq = q0 + q1;
#pragma unroll
      for (int o = 1; o < 16; o <<= 1) {
        rs += __shfl_xor(rs, o);
        rq += __shfl_xor(rq, o);
      }
      lp[r] = lp[r] * al + rs; mp[r] = mpn;
      ln[r] = ln[r] * be + rq; mn[r] = mnn;
      aP0[r] *= al; aP1[r] *= al;
      aN0[r] *= be; aN1[r] *= be;
      const int rr = (r + 8 * hf) * 32;                // C-layout -> A-layout via LDS
      pp[rr + n_lo] = (_Float16)p0;
      pp[rr + 16 + n_lo] = (_Float16)p1;
      pn[rr + n_lo] = (_Float16)q0;
      pn[rr + 16 + n_lo] = (_Float16)q1;
    }
    v16h pa = load_a_frag(pp, 32, lane);               // wave-private: DS in-order
    aP0 = wmma_f16(pa, load_b_frag(VcT, 40, lane), aP0);
    aP1 = wmma_f16(pa, load_b_frag(VcT + 16 * 40, 40, lane), aP1);
    v16h na = load_a_frag(pn, 32, lane);
    aN0 = wmma_f16(na, load_b_frag(VdT, 40, lane), aN0);
    aN1 = wmma_f16(na, load_b_frag(VdT + 16 * 40, 40, lane), aN1);
  }
#pragma unroll
  for (int r = 0; r < 8; ++r) {
    const int t = qt * 64 + wv * 16 + r + 8 * hf;
    _Float16* dst = OP + ((size_t)t * 4 + b) * 1024 + h * 64;
    const float ip = 1.0f / lp[r], id = 1.0f / ln[r];
    dst[n_lo]      = (_Float16)(aP0[r] * ip);          // corr half  (dims 0..31)
    dst[16 + n_lo] = (_Float16)(aP1[r] * ip);
    dst[32 + n_lo] = (_Float16)(aN0[r] * id);          // decorr half (dims 32..63)
    dst[48 + n_lo] = (_Float16)(aN1[r] * id);
  }
}

// ---------------------------------------------------------------------------
// Kernel 4: OUT(f32)[4096,1024] = attn_pre(f16) @ out_w^T + out_b
// B[k,n] = out_w[n,k] -> [n][k] LDS tile = rows of out_w (k contiguous, f32->f16).
// A tile is pure f16 -> async staging. K-tile 64.
__global__ __launch_bounds__(128) void out_gemm_kernel(
    const _Float16* __restrict__ A, const float* __restrict__ OW,
    const float* __restrict__ OB, float* __restrict__ OUT) {
  __shared__ __align__(16) _Float16 As[64 * 72];
  __shared__ __align__(16) _Float16 Bs[64 * 72];
  const int tid = threadIdx.x, lane = tid & 31, wv = tid >> 5;
  const int row0 = blockIdx.x * 64, col0 = blockIdx.y * 64;
  const unsigned long long ab = (unsigned long long)(uintptr_t)A;
  const unsigned as_base = (unsigned)(uintptr_t)As;
  v8f acc[4] = {};
  for (int k0 = 0; k0 < 1024; k0 += 64) {
#pragma unroll 1
    for (int idx = tid; idx < 64 * 8; idx += 128) {   // A tile: async f16 copies
      int r = idx >> 3, c = (idx & 7) * 8;
      unsigned va = (unsigned)((((row0 + r) * 1024) + k0 + c) * 2);
      async_ld_b128(as_base + (unsigned)((r * 72 + c) * 2), va, ab);
    }
#pragma unroll 1
    for (int idx = tid; idx < 64 * 16; idx += 128) {  // B tile: f32 -> f16
      int n = idx >> 4, c = (idx & 15) * 4;
      const float4 v = *(const float4*)(OW + (size_t)(col0 + n) * 1024 + k0 + c);
      Bs[n * 72 + c + 0] = (_Float16)v.x;
      Bs[n * 72 + c + 1] = (_Float16)v.y;
      Bs[n * 72 + c + 2] = (_Float16)v.z;
      Bs[n * 72 + c + 3] = (_Float16)v.w;
    }
    wait_async0();
    __syncthreads();
#pragma unroll
    for (int d = 0; d < 64; d += 32) {
      v16h af = load_a_frag(As + wv * 16 * 72 + d, 72, lane);
#pragma unroll
      for (int nt = 0; nt < 4; ++nt)
        acc[nt] = wmma_f16(af, load_b_frag(Bs + nt * 16 * 72 + d, 72, lane), acc[nt]);
    }
    __syncthreads();
  }
  const int n_lo = lane & 15, hf = lane >> 4;
#pragma unroll
  for (int nt = 0; nt < 4; ++nt) {
    const int col = col0 + nt * 16 + n_lo;
    const float bv = OB[col];
#pragma unroll
    for (int r = 0; r < 8; ++r) {
      const int row = row0 + wv * 16 + r + 8 * hf;
      OUT[(size_t)row * 1024 + col] = acc[nt][r] + bv;
    }
  }
}

// ---------------------------------------------------------------------------
extern "C" void kernel_launch(void* const* d_in, const int* in_sizes, int n_in,
                              void* d_out, int out_size, void* d_ws, size_t ws_size,
                              hipStream_t stream) {
  const float* query = (const float*)d_in[0];   // (T,B,E)
  const float* key   = (const float*)d_in[1];   // (S,B,E)
  const float* value = (const float*)d_in[2];   // (S,B,E)
  const float* W     = (const float*)d_in[3];   // (3E,E)
  const float* bias  = (const float*)d_in[4];   // (3E)
  const float* out_w = (const float*)d_in[5];   // (E,E)
  const float* out_b = (const float*)d_in[6];   // (E)
  float* attn_out = (float*)d_out;                       // 4M f32
  float* avg_out  = (float*)d_out + (size_t)4096 * 1024; // 4M f32

  // Workspace: 4 x (4096x1024) f16 = 32 MB
  _Float16* qf = (_Float16*)d_ws;
  _Float16* kf = qf + (size_t)4096 * 1024;
  _Float16* vf = kf + (size_t)4096 * 1024;
  _Float16* ap = vf + (size_t)4096 * 1024;

  dim3 blk(128);
  // 1) projections (q carries HD^-0.5 = 0.125 scaling)
  proj_gemm_kernel<<<dim3(64, 16), blk, 0, stream>>>(query, W, bias, qf, 0.125f);
  proj_gemm_kernel<<<dim3(64, 16), blk, 0, stream>>>(key, W + 1024 * 1024, bias + 1024, kf, 1.0f);
  proj_gemm_kernel<<<dim3(64, 16), blk, 0, stream>>>(value, W + 2 * 1024 * 1024, bias + 2048, vf, 1.0f);
  // 2) avg_w = (Q_scaled @ K^T)/H per batch (head-sum collapses to full-E dot)
  avgw_gemm_kernel<<<dim3(16, 16, 4), blk, 0, stream>>>(qf, kf, avg_out);
  // 3) dual-softmax flash attention -> attn_pre (f16)
  attn_kernel<<<dim3(16, 64), blk, 0, stream>>>(qf, kf, vf, ap);
  // 4) output projection -> d_out
  out_gemm_kernel<<<dim3(64, 16), blk, 0, stream>>>(ap, out_w, out_b, attn_out);
}
